// LoMoELinearHead_35141422415914
// MI455X (gfx1250) — matrix operand order repaired and checked by reading, not compiled, not run on hardware
//
#include <hip/hip_runtime.h>
#include <hip/hip_bf16.h>

typedef __attribute__((ext_vector_type(16))) __bf16 v16bf;
typedef __attribute__((ext_vector_type(8)))  float  v8f;

#define B_   32
#define L_   2048
#define D_   512
#define O_   512
#define E_   4
#define R_   8
#define H_   256
#define SCALING_ 2.0f

// ---------------------------------------------------------------------------
// 1) x fp32 -> bf16 (vectorized)
// ---------------------------------------------------------------------------
__global__ __launch_bounds__(256) void lomoe_convert_bf16(
    const float* __restrict__ in, __bf16* __restrict__ out, int n4)
{
    int i = blockIdx.x * blockDim.x + threadIdx.x;
    if (i >= n4) return;
    const float4 f = ((const float4*)in)[i];
    union { __bf16 h[4]; uint2 u; } o;
    o.h[0] = (__bf16)f.x; o.h[1] = (__bf16)f.y;
    o.h[2] = (__bf16)f.z; o.h[3] = (__bf16)f.w;
    ((uint2*)out)[i] = o.u;
}

// ---------------------------------------------------------------------------
// 2) mean-pool over L with parallel partial sums + f32 global atomics
//    grid = (16 chunks, B), 256 threads; thread t covers columns t and t+256
// ---------------------------------------------------------------------------
__global__ __launch_bounds__(256) void lomoe_pool(
    const float* __restrict__ x, float* __restrict__ pooled)
{
    const int b = blockIdx.y;
    const int chunk = blockIdx.x;          // 16 chunks of 128 rows
    const int t = threadIdx.x;
    const float* xb = x + ((size_t)b * L_ + (size_t)chunk * 128) * D_;
    float s0 = 0.f, s1 = 0.f;
    for (int l = 0; l < 128; ++l) {
        s0 += xb[(size_t)l * D_ + t];
        s1 += xb[(size_t)l * D_ + t + 256];
    }
    atomicAdd(&pooled[b * D_ + t],       s0);
    atomicAdd(&pooled[b * D_ + t + 256], s1);
}

// ---------------------------------------------------------------------------
// 3) router: MLP -> softmax -> top-2 -> renorm; one block per batch
//    writes probs to d_out tail and SCALING-folded dense weights to ws
// ---------------------------------------------------------------------------
__global__ __launch_bounds__(256) void lomoe_router(
    const float* __restrict__ pooled,
    const float* __restrict__ Wr1, const float* __restrict__ br1,
    const float* __restrict__ Wr2, const float* __restrict__ br2,
    float* __restrict__ probs_out, float* __restrict__ wsc)
{
    __shared__ float sp[D_];
    __shared__ float sh[H_];
    __shared__ float slog[E_];
    const int b = blockIdx.x;
    const int t = threadIdx.x;

    const float invL = 1.0f / (float)L_;
    sp[t]       = pooled[b * D_ + t]       * invL;
    sp[t + 256] = pooled[b * D_ + t + 256] * invL;
    __syncthreads();

    if (t < H_) {
        const float* w = Wr1 + (size_t)t * D_;
        float acc = br1[t];
        #pragma unroll 8
        for (int d = 0; d < D_; ++d) acc += sp[d] * w[d];
        sh[t] = acc > 0.f ? acc : 0.f;
    }
    __syncthreads();

    if (t < E_) {
        const float* w = Wr2 + (size_t)t * H_;
        float acc = br2[t];
        #pragma unroll 8
        for (int j = 0; j < H_; ++j) acc += sh[j] * w[j];
        slog[t] = acc;
    }
    __syncthreads();

    if (t == 0) {
        float mx = slog[0];
        for (int e = 1; e < E_; ++e) mx = slog[e] > mx ? slog[e] : mx;
        float p[E_]; float s = 0.f;
        for (int e = 0; e < E_; ++e) { p[e] = __expf(slog[e] - mx); s += p[e]; }
        const float invs = 1.0f / s;
        for (int e = 0; e < E_; ++e) { p[e] *= invs; probs_out[b * E_ + e] = p[e]; }
        // top-2 (first index wins ties, matching lax.top_k)
        int i0 = 0;
        for (int e = 1; e < E_; ++e) if (p[e] > p[i0]) i0 = e;
        int i1 = -1;
        for (int e = 0; e < E_; ++e) {
            if (e == i0) continue;
            if (i1 < 0 || p[e] > p[i1]) i1 = e;
        }
        float sum2 = p[i0] + p[i1];
        sum2 = sum2 < 1e-6f ? 1e-6f : sum2;
        for (int e = 0; e < E_; ++e) wsc[b * E_ + e] = 0.f;
        wsc[b * E_ + i0] = (p[i0] / sum2) * SCALING_;
        wsc[b * E_ + i1] = (p[i1] / sum2) * SCALING_;
    }
}

// ---------------------------------------------------------------------------
// 4) fold LoRA deltas into per-batch weight matrix:
//    W_eff[b] = bf16(W_base + sum_e wsc[b,e] * B_e @ A_e)
// ---------------------------------------------------------------------------
__global__ __launch_bounds__(256) void lomoe_fold(
    const float* __restrict__ Wb, const float* __restrict__ lA,
    const float* __restrict__ lB, const float* __restrict__ wsc,
    __bf16* __restrict__ weff)
{
    const int idx = blockIdx.x * 256 + threadIdx.x;    // < B*O*D
    const int b   = idx >> 18;                          // / (O_*D_)
    const int rem = idx & (O_ * D_ - 1);
    const int o   = rem >> 9;                           // / D_
    const int d   = rem & (D_ - 1);
    float acc = Wb[rem];
    #pragma unroll
    for (int e = 0; e < E_; ++e) {
        const float we = wsc[b * E_ + e];
        if (we != 0.f) {
            const float* Bp = lB + ((size_t)e * O_ + o) * R_;
            const float* Ap = lA + (size_t)e * R_ * D_ + d;
            float s = 0.f;
            #pragma unroll
            for (int r = 0; r < R_; ++r) s += Bp[r] * Ap[(size_t)r * D_];
            acc += we * s;
        }
    }
    weff[idx] = (__bf16)acc;
}

// ---------------------------------------------------------------------------
// 5) batched WMMA GEMM: out[b] = x_b(bf16) @ W_eff[b]^T + b_base  (f32 acc)
//    grid (L/256, O/128, B); 8 waves/block arranged 4(M) x 2(N);
//    wave tile = 64x64 = 4x4 WMMA tiles -> 16 b128 loads per 16 WMMAs
// ---------------------------------------------------------------------------
__global__ __launch_bounds__(256) void lomoe_gemm(
    const __bf16* __restrict__ xb, const __bf16* __restrict__ wb,
    const float* __restrict__ bias, float* __restrict__ out)
{
    const int b       = blockIdx.z;
    const int row_blk = blockIdx.x * 256;
    const int col_blk = blockIdx.y * 128;
    const int wave = threadIdx.x >> 5;
    const int lane = threadIdx.x & 31;
    const int wm = (wave >> 1) * 64;   // wave row offset within block (0..192)
    const int wn = (wave & 1) * 64;    // wave col offset within block (0,64)
    const int l15 = lane & 15;
    const int hi  = lane >> 4;

    const __bf16* Xb = xb + (size_t)b * ((size_t)L_ * D_);
    const __bf16* Wp = wb + (size_t)b * ((size_t)O_ * D_);

    v8f acc[4][4];
    const v8f vzero = {0.f, 0.f, 0.f, 0.f, 0.f, 0.f, 0.f, 0.f};
    #pragma unroll
    for (int i = 0; i < 4; ++i)
        #pragma unroll
        for (int j = 0; j < 4; ++j) acc[i][j] = vzero;

    // per-lane base pointers (ISA A/B fragment layouts, 16-bit, wave32)
    const __bf16* arow[4];
    #pragma unroll
    for (int i = 0; i < 4; ++i)
        arow[i] = Xb + (size_t)(row_blk + wm + i * 16 + l15) * D_;
    const __bf16* bcol[4];
    #pragma unroll
    for (int j = 0; j < 4; ++j)
        bcol[j] = Wp + (size_t)(col_blk + wn + j * 16 + l15) * D_;

    for (int k0 = 0; k0 < D_; k0 += 32) {
        // unconditional speculative prefetch of next K tile (fault-safe)
        __builtin_prefetch(arow[0] + k0 + 32 + hi * 8, 0, 3);
        __builtin_prefetch(bcol[0] + k0 + 32 + hi * 16, 0, 3);

        v16bf Af[4], Bf[4];
        #pragma unroll
        for (int i = 0; i < 4; ++i) {
            // A 16x32: vgpr0-7 <- K = k0+hi*16 .. +15 (contiguous along D)
            union { v16bf v; uint4 q[2]; } u;
            u.q[0] = *(const uint4*)(arow[i] + k0 + hi * 16);
            u.q[1] = *(const uint4*)(arow[i] + k0 + hi * 16 + 8);
            Af[i] = u.v;
        }
        #pragma unroll
        for (int j = 0; j < 4; ++j) {
            // B 32x16: vgpr0-7 <- K = k0+hi*16 .. +15 (contiguous along D)
            union { v16bf v; uint4 q[2]; } u;
            const uint4* p = (const uint4*)(bcol[j] + k0 + hi * 16);
            u.q[0] = p[0]; u.q[1] = p[1];
            Bf[j] = u.v;
        }
        #pragma unroll
        for (int i = 0; i < 4; ++i)
            #pragma unroll
            for (int j = 0; j < 4; ++j)
                acc[i][j] = __builtin_amdgcn_wmma_f32_16x16x32_bf16(
                    false, Af[i], false, Bf[j], (short)0, acc[i][j], false, false);
    }

    // epilogue: C/D layout row = hi*8 + vgpr, col = lane&15 ; add bias, store f32
    #pragma unroll
    for (int i = 0; i < 4; ++i) {
        const int mbase = row_blk + wm + i * 16 + hi * 8;
        #pragma unroll
        for (int j = 0; j < 4; ++j) {
            const int col = col_blk + wn + j * 16 + l15;
            const float bv = bias[col];
            float* op = out + ((size_t)b * L_ + mbase) * O_ + col;
            #pragma unroll
            for (int v = 0; v < 8; ++v)
                op[(size_t)v * O_] = acc[i][j][v] + bv;
        }
    }
}

// ---------------------------------------------------------------------------
extern "C" void kernel_launch(void* const* d_in, const int* in_sizes, int n_in,
                              void* d_out, int out_size, void* d_ws, size_t ws_size,
                              hipStream_t stream) {
    const float* x      = (const float*)d_in[0];
    const float* W_base = (const float*)d_in[1];
    const float* b_base = (const float*)d_in[2];
    const float* lora_A = (const float*)d_in[3];
    const float* lora_B = (const float*)d_in[4];
    const float* Wr1    = (const float*)d_in[5];
    const float* br1    = (const float*)d_in[6];
    const float* Wr2    = (const float*)d_in[7];
    const float* br2    = (const float*)d_in[8];
    float* out = (float*)d_out;

    // workspace layout
    char* ws = (char*)d_ws;
    const size_t xb_bytes   = (size_t)B_ * L_ * D_ * 2;   // 67,108,864
    const size_t weff_bytes = (size_t)B_ * O_ * D_ * 2;   // 16,777,216
    __bf16* xbuf   = (__bf16*)ws;
    __bf16* weff   = (__bf16*)(ws + xb_bytes);
    float*  pooled = (float*)(ws + xb_bytes + weff_bytes);
    float*  wsc    = (float*)(ws + xb_bytes + weff_bytes + (size_t)B_ * D_ * 4);

    hipMemsetAsync(pooled, 0, (size_t)B_ * D_ * sizeof(float), stream);

    const int n4 = (B_ * L_ * D_) / 4;                    // 8,388,608
    lomoe_convert_bf16<<<n4 / 256, 256, 0, stream>>>(x, xbuf, n4);
    lomoe_pool<<<dim3(16, B_), 256, 0, stream>>>(x, pooled);
    lomoe_router<<<B_, 256, 0, stream>>>(pooled, Wr1, br1, Wr2, br2,
                                         out + (size_t)B_ * L_ * O_, wsc);
    lomoe_fold<<<(B_ * O_ * D_) / 256, 256, 0, stream>>>(W_base, lora_A, lora_B,
                                                         wsc, weff);
    lomoe_gemm<<<dim3(L_ / 256, O_ / 128, B_), 256, 0, stream>>>(xbuf, weff,
                                                                 b_base, out);
}
